// Gemma4MoE_23210003268330
// MI455X (gfx1250) — compile-verified, matrix operand order checked by reading, not run on hardware
//
#include <hip/hip_runtime.h>
#include <math.h>

// Problem constants (from reference)
#define NT 8192   // tokens
#define NH 2048   // hidden
#define NE 32     // experts
#define NI 1024   // intermediate
#define TOPK 2

// Tiling
#define TM 128            // token tile (rows) per block
#define TN 64             // output-col tile per block

typedef __attribute__((ext_vector_type(16))) __bf16 v16bf;
typedef __attribute__((ext_vector_type(8)))  float  v8f;

union BfFrag { uint4 u[2]; v16bf v; };

__device__ __forceinline__ unsigned short f2bf(float x) {
    unsigned int u = __builtin_bit_cast(unsigned int, x);
    unsigned int r = (u + 0x7FFFu + ((u >> 16) & 1u)) >> 16;
    return (unsigned short)r;
}

// Build a v16bf WMMA fragment from two 16B chunks (ISA 16-bit 16x32 layout:
// lane L: row = L%16, chunk0 = K[(L/16)*8 ..+8), chunk1 = K[16+(L/16)*8 ..+8))
__device__ __forceinline__ v16bf load_frag_g(const unsigned short* lo,
                                             const unsigned short* hi) {
    BfFrag f;
    f.u[0] = *(const uint4*)lo;
    f.u[1] = *(const uint4*)hi;
    return f.v;
}

// ---------------------------------------------------------------- init
__global__ __launch_bounds__(256) void init_kernel(float* out, int* counts,
                                                   unsigned short* zrow) {
    size_t i = (size_t)blockIdx.x * blockDim.x + threadIdx.x;
    size_t stride = (size_t)gridDim.x * blockDim.x;
    for (size_t p = i; p < (size_t)NT * NH; p += stride) out[p] = 0.0f;
    if (blockIdx.x == 0 && threadIdx.x < NE) counts[threadIdx.x] = 0;
    if (blockIdx.x == 1) {
        for (int k = threadIdx.x; k < NH; k += 256) zrow[k] = 0;
    }
}

// ------------------------------------------------------ f32 -> bf16 X
__global__ __launch_bounds__(256) void convert_kernel(const float* __restrict__ x,
                                                      unsigned short* __restrict__ xbf) {
    size_t i = (size_t)blockIdx.x * blockDim.x + threadIdx.x;
    size_t stride = (size_t)gridDim.x * blockDim.x;
    for (size_t p = i; p < (size_t)NT * NH; p += stride) xbf[p] = f2bf(x[p]);
}

// ------------------------------------- weights: [E][K][N] f32 -> [E][N][K] bf16
__global__ __launch_bounds__(256)
void tconv_kernel(const float* __restrict__ src, unsigned short* __restrict__ dst,
                  int K, int N) {
    const int e = blockIdx.z;
    const int k0 = blockIdx.y * 64;
    const int n0 = blockIdx.x * 64;
    __shared__ unsigned short t_lds[64][72];   // [n][k], padded
    const int tid = threadIdx.x;
    // read 64(k) x 64(n) f32 tile, coalesced along n, store transposed bf16
#pragma unroll
    for (int i = 0; i < 4; ++i) {
        int seg = tid + i * 256;            // 0..1023, 16 float4 per k-row
        int k = seg >> 4;
        int n4 = (seg & 15) * 4;
        float4 v = *(const float4*)(src + ((size_t)e * K + k0 + k) * N + n0 + n4);
        t_lds[n4 + 0][k] = f2bf(v.x);
        t_lds[n4 + 1][k] = f2bf(v.y);
        t_lds[n4 + 2][k] = f2bf(v.z);
        t_lds[n4 + 3][k] = f2bf(v.w);
    }
    __syncthreads();
    // write 64(n) rows of 64 bf16, coalesced 16B segments
#pragma unroll
    for (int i = 0; i < 2; ++i) {
        int seg = tid + i * 256;            // 0..511, 8 segs per n-row
        int n = seg >> 3;
        int k8 = (seg & 7) * 8;
        *(uint4*)(dst + ((size_t)e * N + n0 + n) * K + k0 + k8) =
            *(const uint4*)&t_lds[n][k8];
    }
}

// ---------------------------------------------------------------- routing
__global__ __launch_bounds__(256) void routing_kernel(const float* __restrict__ logits,
                                                      const float* __restrict__ scale,
                                                      int* __restrict__ counts,
                                                      int* __restrict__ tok_ids,
                                                      float* __restrict__ tok_w) {
    int t = blockIdx.x * blockDim.x + threadIdx.x;
    if (t >= NT) return;
    const float* row = logits + (size_t)t * NE;
    float b1 = -1e30f, b2 = -1e30f;
    int i1 = 0, i2 = 0;
    for (int e = 0; e < NE; ++e) {
        float v = row[e];
        if (v > b1) { b2 = b1; i2 = i1; b1 = v; i1 = e; }
        else if (v > b2) { b2 = v; i2 = e; }
    }
    float e2 = __expf(b2 - b1);
    float inv = 1.0f / (1.0f + e2);
    float w1 = inv * scale[i1];
    float w2 = e2 * inv * scale[i2];
    tok_ids[t * 2 + 0] = i1;  tok_w[t * 2 + 0] = w1;
    tok_ids[t * 2 + 1] = i2;  tok_w[t * 2 + 1] = w2;
    atomicAdd(&counts[i1], 1);
    atomicAdd(&counts[i2], 1);
}

// ---------------------------------------------------------------- scan
__global__ void scan_kernel(const int* __restrict__ counts,
                            int* __restrict__ offsets,
                            int* __restrict__ cursors) {
    if (threadIdx.x == 0) {
        int run = 0;
        for (int e = 0; e < NE; ++e) {
            offsets[e] = run;
            cursors[e] = run;
            run += counts[e];
        }
    }
}

// ---------------------------------------------------------------- scatter
__global__ __launch_bounds__(256) void scatter_kernel(const int* __restrict__ tok_ids,
                                                      const float* __restrict__ tok_w,
                                                      int* __restrict__ cursors,
                                                      int* __restrict__ btok,
                                                      float* __restrict__ bw) {
    int t = blockIdx.x * blockDim.x + threadIdx.x;
    if (t >= NT) return;
    for (int j = 0; j < TOPK; ++j) {
        int id = tok_ids[t * 2 + j];
        int slot = atomicAdd(&cursors[id], 1);
        btok[slot] = t;
        bw[slot] = tok_w[t * 2 + j];
    }
}

// ------------------------------------------------- gate/up projection + GELU
// A = xbf rows (gathered tokens), B = wgt/wut [E][NI][NH] (K-contiguous bf16).
// Barrier-free, LDS-free inner loop: fragments straight from global.
__global__ __launch_bounds__(256)
void moe_gateup_kernel(const unsigned short* __restrict__ wgt,
                       const unsigned short* __restrict__ wut,
                       const unsigned short* __restrict__ xbf,
                       const unsigned short* __restrict__ zrow,
                       unsigned short* __restrict__ hbuf,
                       const int* __restrict__ counts, const int* __restrict__ offsets,
                       const int* __restrict__ btok, const float* __restrict__ bw) {
    const int e = blockIdx.z;
    const int mtile = blockIdx.y;
    const int ntile = blockIdx.x;
    const int cnt = counts[e];
    if (mtile * TM >= cnt) return;
    const int offs = offsets[e];
    const int nbase = ntile * TN;

    __shared__ float w_lds[TM];

    const int tid = threadIdx.x;
    const int lane = tid & 31;
    const int wv = tid >> 5;
    const int wm = wv >> 1;       // 0..3
    const int wn = wv & 1;        // 0..1
    const int half = lane >> 4;
    const int l16 = lane & 15;

    if (tid < TM) {
        int gr = mtile * TM + tid;
        w_lds[tid] = (gr < cnt) ? bw[offs + gr] : 0.0f;
    }

    // per-lane operand row base pointers
    const unsigned short* aptr[2];
#pragma unroll
    for (int ti = 0; ti < 2; ++ti) {
        int grow = mtile * TM + wm * 32 + ti * 16 + l16;
        aptr[ti] = (grow < cnt) ? (xbf + (size_t)btok[offs + grow] * NH) : zrow;
    }
    const unsigned short* gptr[2];
    const unsigned short* uptr[2];
#pragma unroll
    for (int tj = 0; tj < 2; ++tj) {
        int n = nbase + wn * 32 + tj * 16 + l16;
        gptr[tj] = wgt + ((size_t)e * NI + n) * NH;
        uptr[tj] = wut + ((size_t)e * NI + n) * NH;
    }

    v8f acc_g[2][2] = {};
    v8f acc_u[2][2] = {};

#pragma unroll 2
    for (int k0 = 0; k0 < NH; k0 += 32) {
        const int klo = k0 + half * 8;
        const int khi = k0 + 16 + half * 8;
        __builtin_prefetch(aptr[0] + k0 + 512, 0, 0);
        __builtin_prefetch(gptr[0] + k0 + 512, 0, 0);
        __builtin_prefetch(uptr[0] + k0 + 512, 0, 0);
        v16bf a[2], bg[2], bu[2];
#pragma unroll
        for (int ti = 0; ti < 2; ++ti)
            a[ti] = load_frag_g(aptr[ti] + klo, aptr[ti] + khi);
#pragma unroll
        for (int tj = 0; tj < 2; ++tj) {
            bg[tj] = load_frag_g(gptr[tj] + klo, gptr[tj] + khi);
            bu[tj] = load_frag_g(uptr[tj] + klo, uptr[tj] + khi);
        }
#pragma unroll
        for (int ti = 0; ti < 2; ++ti)
#pragma unroll
            for (int tj = 0; tj < 2; ++tj) {
                acc_g[ti][tj] = __builtin_amdgcn_wmma_f32_16x16x32_bf16(
                    false, a[ti], false, bg[tj], (short)0, acc_g[ti][tj], false, false);
                acc_u[ti][tj] = __builtin_amdgcn_wmma_f32_16x16x32_bf16(
                    false, a[ti], false, bu[tj], (short)0, acc_u[ti][tj], false, false);
            }
    }

    __syncthreads();   // w_lds visible
    // epilogue: h = gelu(gate) * up * combine_weight -> bf16 hbuf
#pragma unroll
    for (int ti = 0; ti < 2; ++ti)
#pragma unroll
        for (int tj = 0; tj < 2; ++tj)
#pragma unroll
            for (int v = 0; v < 8; ++v) {
                int m = v + 8 * half;
                int lrow = wm * 32 + ti * 16 + m;
                int grow = mtile * TM + lrow;
                if (grow < cnt) {
                    float g = acc_g[ti][tj][v];
                    float u = acc_u[ti][tj][v];
                    float hval = 0.5f * g * (1.0f + erff(g * 0.70710678118654752f));
                    hval = hval * u * w_lds[lrow];
                    int col = nbase + wn * 32 + tj * 16 + l16;
                    hbuf[(size_t)(offs + grow) * NI + col] = f2bf(hval);
                }
            }
}

// ------------------------------------------------- down projection + combine
__global__ __launch_bounds__(256)
void moe_down_kernel(const unsigned short* __restrict__ wdt,   // [E][NH][NI]
                     const unsigned short* __restrict__ hbuf,  // [NT*K][NI]
                     const unsigned short* __restrict__ zrow,
                     float* __restrict__ out,
                     const int* __restrict__ counts, const int* __restrict__ offsets,
                     const int* __restrict__ btok) {
    const int e = blockIdx.z;
    const int mtile = blockIdx.y;
    const int ntile = blockIdx.x;
    const int cnt = counts[e];
    if (mtile * TM >= cnt) return;
    const int offs = offsets[e];
    const int nbase = ntile * TN;

    __shared__ int tok_lds[TM];

    const int tid = threadIdx.x;
    const int lane = tid & 31;
    const int wv = tid >> 5;
    const int wm = wv >> 1;
    const int wn = wv & 1;
    const int half = lane >> 4;
    const int l16 = lane & 15;

    if (tid < TM) {
        int gr = mtile * TM + tid;
        tok_lds[tid] = (gr < cnt) ? btok[offs + gr] : 0;
    }

    const unsigned short* aptr[2];
#pragma unroll
    for (int ti = 0; ti < 2; ++ti) {
        int grow = mtile * TM + wm * 32 + ti * 16 + l16;
        aptr[ti] = (grow < cnt) ? (hbuf + (size_t)(offs + grow) * NI) : zrow;
    }
    const unsigned short* bptr[2];
#pragma unroll
    for (int tj = 0; tj < 2; ++tj) {
        int n = nbase + wn * 32 + tj * 16 + l16;
        bptr[tj] = wdt + ((size_t)e * NH + n) * NI;
    }

    v8f acc[2][2] = {};

#pragma unroll 2
    for (int k0 = 0; k0 < NI; k0 += 32) {
        const int klo = k0 + half * 8;
        const int khi = k0 + 16 + half * 8;
        __builtin_prefetch(aptr[0] + k0 + 512, 0, 0);
        __builtin_prefetch(bptr[0] + k0 + 512, 0, 0);
        v16bf a[2], b[2];
#pragma unroll
        for (int ti = 0; ti < 2; ++ti)
            a[ti] = load_frag_g(aptr[ti] + klo, aptr[ti] + khi);
#pragma unroll
        for (int tj = 0; tj < 2; ++tj)
            b[tj] = load_frag_g(bptr[tj] + klo, bptr[tj] + khi);
#pragma unroll
        for (int ti = 0; ti < 2; ++ti)
#pragma unroll
            for (int tj = 0; tj < 2; ++tj)
                acc[ti][tj] = __builtin_amdgcn_wmma_f32_16x16x32_bf16(
                    false, a[ti], false, b[tj], (short)0, acc[ti][tj], false, false);
    }

    __syncthreads();   // tok_lds visible
#pragma unroll
    for (int ti = 0; ti < 2; ++ti)
#pragma unroll
        for (int tj = 0; tj < 2; ++tj)
#pragma unroll
            for (int v = 0; v < 8; ++v) {
                int m = v + 8 * half;
                int lrow = wm * 32 + ti * 16 + m;
                int grow = mtile * TM + lrow;
                if (grow < cnt) {
                    int tok = tok_lds[lrow];
                    int col = nbase + wn * 32 + tj * 16 + l16;
                    atomicAdd(&out[(size_t)tok * NH + col], acc[ti][tj][v]);
                }
            }
}

// ---------------------------------------------------------------- launch
extern "C" void kernel_launch(void* const* d_in, const int* in_sizes, int n_in,
                              void* d_out, int out_size, void* d_ws, size_t ws_size,
                              hipStream_t stream) {
    const float* hs     = (const float*)d_in[0];  // [T,H]
    const float* logits = (const float*)d_in[1];  // [T,E]
    const float* scale  = (const float*)d_in[2];  // [E]
    const float* wg     = (const float*)d_in[3];  // [E,H,I]
    const float* wu     = (const float*)d_in[4];  // [E,H,I]
    const float* wd     = (const float*)d_in[5];  // [E,I,H]
    float* out = (float*)d_out;

    // workspace carve-up (16B aligned)
    uintptr_t p = (uintptr_t)d_ws;
    p = (p + 15) & ~(uintptr_t)15;
    int* counts  = (int*)p;            p += NE * sizeof(int);
    int* offsets = (int*)p;            p += NE * sizeof(int);
    int* cursors = (int*)p;            p += NE * sizeof(int);
    int* btok    = (int*)p;            p += (size_t)NT * TOPK * sizeof(int);
    float* bw    = (float*)p;          p += (size_t)NT * TOPK * sizeof(float);
    int* tok_ids = (int*)p;            p += (size_t)NT * TOPK * sizeof(int);
    float* tok_w = (float*)p;          p += (size_t)NT * TOPK * sizeof(float);
    p = (p + 255) & ~(uintptr_t)255;
    unsigned short* zrow = (unsigned short*)p;  p += (size_t)NH * sizeof(unsigned short);
    p = (p + 255) & ~(uintptr_t)255;
    unsigned short* xbf  = (unsigned short*)p;  p += (size_t)NT * NH * sizeof(unsigned short);
    unsigned short* hbuf = (unsigned short*)p;  p += (size_t)NT * TOPK * NI * sizeof(unsigned short);
    unsigned short* wgt  = (unsigned short*)p;  p += (size_t)NE * NH * NI * sizeof(unsigned short);
    unsigned short* wut  = (unsigned short*)p;  p += (size_t)NE * NH * NI * sizeof(unsigned short);
    unsigned short* wdt  = (unsigned short*)p;  // [E][NH][NI]

    init_kernel<<<4096, 256, 0, stream>>>(out, counts, zrow);
    convert_kernel<<<4096, 256, 0, stream>>>(hs, xbf);
    routing_kernel<<<NT / 256, 256, 0, stream>>>(logits, scale, counts, tok_ids, tok_w);
    scan_kernel<<<1, 32, 0, stream>>>(counts, offsets, cursors);
    scatter_kernel<<<NT / 256, 256, 0, stream>>>(tok_ids, tok_w, cursors, btok, bw);

    dim3 gT1(NI / 64, NH / 64, NE);  // wg, wu: K=NH, N=NI
    tconv_kernel<<<gT1, 256, 0, stream>>>(wg, wgt, NH, NI);
    tconv_kernel<<<gT1, 256, 0, stream>>>(wu, wut, NH, NI);
    dim3 gT2(NH / 64, NI / 64, NE);  // wd: K=NI, N=NH
    tconv_kernel<<<gT2, 256, 0, stream>>>(wd, wdt, NI, NH);

    dim3 gA(NI / TN, NT / TM, NE);   // (16, 64, 32), inactive tiles early-exit
    moe_gateup_kernel<<<gA, 256, 0, stream>>>(wgt, wut, xbf, zrow, hbuf,
                                              counts, offsets, btok, bw);
    dim3 gB(NH / TN, NT / TM, NE);   // (32, 64, 32)
    moe_down_kernel<<<gB, 256, 0, stream>>>(wdt, hbuf, zrow, out,
                                            counts, offsets, btok);
}
